// Attend_21174188769972
// MI455X (gfx1250) — matrix operand order repaired and checked by reading, not compiled, not run on hardware
//
#include <hip/hip_runtime.h>

// B=2, H=16, N=2048, D=64, TOPK=64
#define Bsz 2
#define Hn  16
#define Nn  2048
#define Dd  64
#define TOPK 64

typedef __attribute__((ext_vector_type(16))) __bf16 v16bf;
typedef __attribute__((ext_vector_type(8)))  float  v8f;
typedef __attribute__((ext_vector_type(4)))  unsigned v4u;
typedef __attribute__((ext_vector_type(8)))  int   v8i;
typedef __attribute__((ext_vector_type(4)))  int   v4i;

// ---- toolchain probes (reported via stderr even on successful compile) --
#if __has_builtin(__builtin_amdgcn_tensor_load_to_lds)
#warning CDNA5_PROBE: tensor_load_to_lds builtin PRESENT
#else
#warning CDNA5_PROBE: tensor_load_to_lds builtin ABSENT - using inline asm
#endif
#if __has_builtin(__builtin_amdgcn_s_wait_asynccnt)
#warning CDNA5_PROBE: s_wait_asynccnt builtin PRESENT
#else
#warning CDNA5_PROBE: s_wait_asynccnt builtin ABSENT - using inline asm
#endif

// ---- CDNA5 async global->LDS copy (ASYNCcnt-tracked) -------------------
__device__ __forceinline__ void async_copy_b128(unsigned lds_off,
                                                unsigned long long gaddr) {
  asm volatile("global_load_async_to_lds_b128 %0, %1, off"
               :: "v"(lds_off), "v"(gaddr) : "memory");
}
__device__ __forceinline__ void wait_asynccnt0() {
#if __has_builtin(__builtin_amdgcn_s_wait_asynccnt)
  __builtin_amdgcn_s_wait_asynccnt(0);
#else
  asm volatile("s_wait_asynccnt 0x0" ::: "memory");
#endif
}

// ---- CDNA5 Tensor Data Mover: 2D f32 tile load (TENSORcnt-tracked) -----
// D# per ISA 8.3/8.4: group0 = {flags, lds_addr, global_addr, type=2},
// group1 = {data_size=4B, tensor_dim0/1, tile_dim0/1, dim0_stride}.
__device__ __forceinline__ void tdm_load_2d_f32(unsigned lds_addr,
                                                unsigned long long gaddr,
                                                int tile_rows, int tile_cols,
                                                int tensor_rows, int row_stride) {
  v4u g0;
  g0[0] = 1u;                                           // count=1 (user mode)
  g0[1] = lds_addr;                                     // LDS byte address
  g0[2] = (unsigned)(gaddr & 0xFFFFFFFFu);              // global_addr[31:0]
  g0[3] = (unsigned)((gaddr >> 32) & 0x01FFFFFFu)       // global_addr[56:32]
          | (2u << 30);                                 // type=2 ("image")
  v8i g1;
  g1[0] = (int)(2u << 16);                              // data_size=2 (4 bytes)
  g1[1] = (int)(((unsigned)tile_cols & 0xFFFFu) << 16); // tensor_dim0[15:0]
  g1[2] = (int)((((unsigned)tile_cols >> 16) & 0xFFFFu)
          | (((unsigned)tensor_rows & 0xFFFFu) << 16)); // dim0 hi | dim1 lo
  g1[3] = (int)((((unsigned)tensor_rows >> 16) & 0xFFFFu)
          | (((unsigned)tile_cols & 0xFFFFu) << 16));   // dim1 hi | tile_dim0
  g1[4] = tile_rows & 0xFFFF;                           // tile_dim1 (tile_dim2=0)
  g1[5] = row_stride;                                   // tensor_dim0_stride lo32
  g1[6] = 0;
  g1[7] = 0;
#if __has_builtin(__builtin_amdgcn_tensor_load_to_lds)
  v4i gz = {0, 0, 0, 0};
#if __clang_major__ >= 23
  v8i gz8 = {0, 0, 0, 0, 0, 0, 0, 0};
  __builtin_amdgcn_tensor_load_to_lds(g0, g1, gz, gz, gz8, 0);
#else
  __builtin_amdgcn_tensor_load_to_lds(g0, g1, gz, gz, 0);
#endif
#else
  // 2-SGPR-group form (VADDR2/VADDR3 disabled): tensors up to 2D
  asm volatile("tensor_load_to_lds %0, %1" :: "s"(g0), "s"(g1) : "memory");
#endif
}
__device__ __forceinline__ void wait_tensorcnt0() {
#if __has_builtin(__builtin_amdgcn_s_wait_tensorcnt)
  __builtin_amdgcn_s_wait_tensorcnt(0);
#else
  asm volatile("s_wait_tensorcnt 0x0" ::: "memory");
#endif
}

// ---- CDNA5 WMMA fragment builders (16x16x32 bf16) ----------------------
template <typename F>
__device__ __forceinline__ v16bf make_a_frag(F f) {
  int lane = threadIdx.x & 31;
  int m = lane & 15;
  int kh = (lane >> 4) << 3;          // +8 for upper half-wave
  v16bf a;
#pragma unroll
  for (int r = 0; r < 8; ++r) {
    int kb = (r < 4 ? (r << 1) : 16 + ((r - 4) << 1)) + kh;
    a[2 * r]     = f(m, kb);
    a[2 * r + 1] = f(m, kb + 1);
  }
  return a;
}
template <typename F>
__device__ __forceinline__ v16bf make_b_frag(F f) {
  int lane = threadIdx.x & 31;
  int n = lane & 15;
  int kh = (lane >> 4) << 3;          // +8 for upper half-wave
  v16bf b;
#pragma unroll
  for (int r = 0; r < 8; ++r) {
    int kb = (r < 4 ? (r << 1) : 16 + ((r - 4) << 1)) + kh;
    b[2 * r]     = f(kb, n);
    b[2 * r + 1] = f(kb + 1, n);
  }
  return b;
}

// ---- Kernel 1: Dmix[b,g,i,j] = sum_h w_pre[g,h]*scale*(q_h . k_h) + bias
__global__ __launch_bounds__(512) void dots_kernel(
    const float* __restrict__ q, const float* __restrict__ k,
    const float* __restrict__ attn_bias, const float* __restrict__ w_pre,
    float* __restrict__ Dmix) {
  __shared__ __align__(16) float qf[Hn][16][Dd];    // 64KB raw f32 q (async)
  __shared__ __align__(16) float kf[Hn][16][Dd];    // 64KB raw f32 k (TDM)
  __shared__ __bf16 qs[Hn][16][Dd];                 // 32KB bf16 q (scaled)
  __shared__ __bf16 ks[Hn][16][Dd];                 // 32KB bf16 k
  __shared__ __bf16 Sp[16][16][32];                 // 16KB [i][j][h(pad32)]
  __shared__ __bf16 wp[16][32];                     // w_pre padded to K=32

  int tid = threadIdx.x, lane = tid & 31, wave = tid >> 5;
  int b = blockIdx.z, it = blockIdx.y, jt = blockIdx.x;
  int i0 = it * 16, j0 = jt * 16;

  // --- TDM: wave h pulls its head's k tile (16 rows x 64 f32) ---
  {
    int h = wave;
    unsigned long long ga = (unsigned long long)(uintptr_t)
        (k + (((size_t)b * Hn + h) * Nn + j0) * Dd);
    unsigned la = (unsigned)(uintptr_t)&kf[h][0][0];
    tdm_load_2d_f32(la, ga, /*tile_rows=*/16, /*tile_cols=*/Dd,
                    /*tensor_rows=*/Nn, /*row_stride=*/Dd);
  }

  // --- async global->LDS staging of the q tile (raw f32, b128 chunks) ---
  unsigned qfbase = (unsigned)(uintptr_t)&qf[0][0][0];
  for (int idx = tid; idx < Hn * 16 * (Dd / 4); idx += 512) {  // 4096 x b128
    int d4 = (idx & 15) << 2;
    int i  = (idx >> 4) & 15;
    int h  = idx >> 8;
    unsigned long long ga = (unsigned long long)(uintptr_t)
        (q + (((size_t)b * Hn + h) * Nn + i0 + i) * Dd + d4);
    async_copy_b128(qfbase + (unsigned)(((h * 16 + i) * Dd + d4) * 4), ga);
  }

  for (int idx = tid; idx < 16 * 32; idx += 512) {
    int hh = idx & 31, g = idx >> 5;
    wp[g][hh] = (hh < 16) ? (__bf16)w_pre[g * Hn + hh] : (__bf16)0.f;
  }
  for (int idx = tid; idx < 16 * 16 * 16; idx += 512) {  // zero h=16..31 pad
    int hh = 16 + (idx & 15), j = (idx >> 4) & 15, i = idx >> 8;
    Sp[i][j][hh] = (__bf16)0.f;
  }
  wait_asynccnt0();            // q async copies landed (this wave)
  wait_tensorcnt0();           // k TDM tile landed (this wave)
  __syncthreads();             // -> all waves' transfers landed

  for (int idx = tid; idx < Hn * 16 * Dd; idx += 512) {  // convert to bf16
    int d = idx & 63, i = (idx >> 6) & 15, h = idx >> 10;
    qs[h][i][d] = (__bf16)(qf[h][i][d] * 0.125f);        // scale = D^-0.5
    ks[h][i][d] = (__bf16)kf[h][i][d];
  }
  __syncthreads();

  {  // wave h: S_h(16i x 16j) = q_h(16x64) * k_h^T(64x16)
    int h = wave;
    v8f acc = {};
#pragma unroll
    for (int kc = 0; kc < 2; ++kc) {
      v16bf a  = make_a_frag([&](int m, int kk) { return qs[h][m][kc * 32 + kk]; });
      v16bf bb = make_b_frag([&](int kk, int n) { return ks[h][n][kc * 32 + kk]; });
      acc = __builtin_amdgcn_wmma_f32_16x16x32_bf16(false, a, false, bb,
                                                    (short)0, acc, false, false);
    }
    int j = lane & 15, mh = (lane >> 4) * 8;
#pragma unroll
    for (int r = 0; r < 8; ++r) Sp[mh + r][j][h] = (__bf16)acc[r];
  }
  __syncthreads();

  {  // wave i: Dg(16g x 16j) = w_pre(16x32h) * Sp[i](32h x 16j), +bias, store
    int i = wave;
    v16bf a  = make_a_frag([&](int m, int kk) { return wp[m][kk]; });
    v16bf bb = make_b_frag([&](int kk, int n) { return Sp[i][n][kk]; });
    v8f c = {};
    c = __builtin_amdgcn_wmma_f32_16x16x32_bf16(false, a, false, bb,
                                                (short)0, c, false, false);
    int j = lane & 15, gh = (lane >> 4) * 8;
#pragma unroll
    for (int r = 0; r < 8; ++r) {
      int g = gh + r;
      float bias = attn_bias[(((size_t)g) * Nn + (i0 + i)) * Nn + j0 + j];
      Dmix[(((size_t)b * Hn + g) * Nn + (i0 + i)) * Nn + j0 + j] = c[r] + bias;
    }
  }
}

// ---- Kernel 2: per-row exact 64th-largest (radix select) + softmax stats
__device__ __forceinline__ float key_to_float(unsigned kk) {
  unsigned u = (kk >> 31) ? (kk & 0x7FFFFFFFu) : ~kk;
  return __uint_as_float(u);
}

__global__ __launch_bounds__(256) void topk_kernel(
    const float* __restrict__ Dmix, float* __restrict__ kth,
    float* __restrict__ mrow, float* __restrict__ sinv) {
  __shared__ int   hist[8][16];
  __shared__ float red[8][32];
  int lane = threadIdx.x & 31, wave = threadIdx.x >> 5;
  size_t row = (size_t)blockIdx.x * 8 + wave;        // < B*H*N = 65536
  const float* src = Dmix + row * (size_t)Nn;

  unsigned key[64];
#pragma unroll
  for (int t = 0; t < 64; ++t) {
    unsigned u = __float_as_uint(src[t * 32 + lane]);
    key[t] = (u >> 31) ? ~u : (u | 0x80000000u);     // order-preserving map
  }

  unsigned prefix = 0;
  int kwant = TOPK;                                  // 64th largest
  for (int p = 0; p < 8; ++p) {
    int shift = 28 - p * 4;
    if (lane < 16) hist[wave][lane] = 0;
    __syncthreads();
#pragma unroll
    for (int t = 0; t < 64; ++t) {
      unsigned kk = key[t];
      bool match = (p == 0) || ((kk >> (shift + 4)) == prefix);
      if (match) atomicAdd(&hist[wave][(kk >> shift) & 15], 1);
    }
    __syncthreads();
    int cum = 0, sel = 0;
    for (int d2 = 15; d2 >= 0; --d2) {               // redundant scan, all lanes
      int c = hist[wave][d2];
      if (cum + c >= kwant) { sel = d2; break; }
      cum += c;
    }
    kwant -= cum;
    prefix = (p == 0) ? (unsigned)sel : ((prefix << 4) | (unsigned)sel);
    __syncthreads();
  }
  float kthf = key_to_float(prefix);

  float mloc = -3.4e38f;
#pragma unroll
  for (int t = 0; t < 64; ++t)
    if (key[t] < prefix) mloc = fmaxf(mloc, key_to_float(key[t]));
  red[wave][lane] = mloc;
  __syncthreads();
  float m = -3.4e38f;
  for (int t = 0; t < 32; ++t) m = fmaxf(m, red[wave][t]);
  __syncthreads();
  float sloc = 0.f;
#pragma unroll
  for (int t = 0; t < 64; ++t)
    if (key[t] < prefix) sloc += __expf(key_to_float(key[t]) - m);
  red[wave][lane] = sloc;
  __syncthreads();
  float s = 0.f;
  for (int t = 0; t < 32; ++t) s += red[wave][t];
  if (lane == 0) { kth[row] = kthf; mrow[row] = m; sinv[row] = 1.f / s; }
}

// ---- Kernel 3: normalize+mask -> w_post head-mix (WMMA) -> attn.v (WMMA)
__global__ __launch_bounds__(512) void out_kernel(
    const float* __restrict__ Dmix, const float* __restrict__ v,
    const float* __restrict__ w_post, const float* __restrict__ kth,
    const float* __restrict__ mrow, const float* __restrict__ sinv,
    float* __restrict__ out) {
  __shared__ __bf16 P[16][32][32];    // attn (normalized) [i][j][h(pad32)] 32KB
  __shared__ __bf16 AP[Hn][16][32];   // post-mixed attn  [g][i][j]         16KB
  __shared__ __bf16 vs[Hn][32][Dd];   // v tile           [g][j][d]         64KB
  __shared__ __bf16 wq[16][32];       // w_post padded to K=32
  __shared__ float kthS[Hn][16], mS[Hn][16], isS[Hn][16];

  int tid = threadIdx.x, lane = tid & 31, wave = tid >> 5;
  int b = blockIdx.y, i0 = blockIdx.x * 16;
  int g = wave;

  for (int idx = tid; idx < 16 * 32; idx += 512) {
    int hh = idx & 31, gg = idx >> 5;
    wq[gg][hh] = (hh < 16) ? (__bf16)w_post[gg * Hn + hh] : (__bf16)0.f;
  }
  for (int idx = tid; idx < Hn * 16; idx += 512) {
    int i = idx & 15, h = idx >> 4;
    size_t r = ((size_t)b * Hn + h) * Nn + i0 + i;
    kthS[h][i] = kth[r]; mS[h][i] = mrow[r]; isS[h][i] = sinv[r];
  }
  for (int idx = tid; idx < 16 * 32 * 16; idx += 512) {  // zero P pad h=16..31
    int hh = 16 + (idx & 15), j = (idx >> 4) & 31, i = idx >> 9;
    P[i][j][hh] = (__bf16)0.f;
  }

  v8f acc[4] = {};                    // 16i x 64d f32 accumulator per wave (g)
  for (int jt = 0; jt < Nn / 32; ++jt) {
    int j0 = jt * 32;
    if (jt + 1 < Nn / 32) {          // prefetch next Dmix j-tile
      int ph = tid >> 5, pi = (tid >> 1) & 15, pj = (tid & 1) * 16;
      __builtin_prefetch(
          &Dmix[(((size_t)b * Hn + ph) * Nn + i0 + pi) * Nn + (jt + 1) * 32 + pj],
          0, 1);
    }
    __syncthreads();
    for (int idx = tid; idx < Hn * 16 * 32; idx += 512) {
      int j = idx & 31, i = (idx >> 5) & 15, h = idx >> 9;
      float x = Dmix[(((size_t)b * Hn + h) * Nn + i0 + i) * Nn + j0 + j];
      float p = (x < kthS[h][i]) ? __expf(x - mS[h][i]) * isS[h][i] : 0.f;
      P[i][j][h] = (__bf16)p;
    }
    for (int idx = tid; idx < Hn * 32 * Dd; idx += 512) {
      int d = idx & 63, j = (idx >> 6) & 31, h = idx >> 11;
      vs[h][j][d] = (__bf16)v[(((size_t)b * Hn + h) * Nn + j0 + j) * Dd + d];
    }
    __syncthreads();
    {  // wave i: AP_i(16g x 32j) = w_post(16x32h) * P[i](32h x 32j)
      int i = wave;
#pragma unroll
      for (int sl = 0; sl < 2; ++sl) {
        v16bf a  = make_a_frag([&](int m, int kk) { return wq[m][kk]; });
        v16bf bb = make_b_frag([&](int kk, int n) { return P[i][sl * 16 + n][kk]; });
        v8f c = {};
        c = __builtin_amdgcn_wmma_f32_16x16x32_bf16(false, a, false, bb,
                                                    (short)0, c, false, false);
        int j = (lane & 15) + sl * 16, gh = (lane >> 4) * 8;
#pragma unroll
        for (int r = 0; r < 8; ++r) AP[gh + r][i][j] = (__bf16)c[r];
      }
    }
    __syncthreads();
    {  // wave g: acc += AP[g](16i x 32j) * vs[g](32j x 64d)
      v16bf a = make_a_frag([&](int m, int kk) { return AP[g][m][kk]; });
#pragma unroll
      for (int dc = 0; dc < 4; ++dc) {
        v16bf bb = make_b_frag([&](int kk, int n) { return vs[g][kk][dc * 16 + n]; });
        acc[dc] = __builtin_amdgcn_wmma_f32_16x16x32_bf16(false, a, false, bb,
                                                          (short)0, acc[dc], false, false);
      }
    }
  }
  {  // store out[b][g][i0+M][d]
    int n = lane & 15, ih = (lane >> 4) * 8;
#pragma unroll
    for (int dc = 0; dc < 4; ++dc)
#pragma unroll
      for (int r = 0; r < 8; ++r)
        out[(((size_t)b * Hn + g) * Nn + i0 + ih + r) * Dd + dc * 16 + n] = acc[dc][r];
  }
}

extern "C" void kernel_launch(void* const* d_in, const int* in_sizes, int n_in,
                              void* d_out, int out_size, void* d_ws, size_t ws_size,
                              hipStream_t stream) {
  const float* q         = (const float*)d_in[0];
  const float* k         = (const float*)d_in[1];
  const float* v         = (const float*)d_in[2];
  const float* attn_bias = (const float*)d_in[3];
  const float* w_pre     = (const float*)d_in[4];
  const float* w_post    = (const float*)d_in[5];
  // d_in[6] = sparse_topk (compile-time TOPK=64)
  float* out = (float*)d_out;

  // workspace: Dmix (B*H*N*N f32 = 512MB) + kth/m/1-over-s (3 * 64K f32)
  char* ws = (char*)d_ws;
  float* Dmix = (float*)ws;
  size_t dmixElems = (size_t)Bsz * Hn * Nn * Nn;
  float* kth  = (float*)(ws + dmixElems * sizeof(float));
  float* mrow = kth + (size_t)Bsz * Hn * Nn;
  float* sinv = mrow + (size_t)Bsz * Hn * Nn;

  dim3 gA(Nn / 16, Nn / 16, Bsz);                       // 128 x 128 x 2
  dots_kernel<<<gA, 512, 0, stream>>>(q, k, attn_bias, w_pre, Dmix);

  int rows = Bsz * Hn * Nn;                             // 65536
  topk_kernel<<<rows / 8, 256, 0, stream>>>(Dmix, kth, mrow, sinv);

  dim3 gC(Nn / 16, Bsz);                                // 128 x 2
  out_kernel<<<gC, 512, 0, stream>>>(Dmix, v, w_post, kth, mrow, sinv, out);
}
